// TinySpikingAutoencoder_27616639713882
// MI455X (gfx1250) — matrix operand order repaired and checked by reading, not compiled, not run on hardware
//
#include <hip/hip_runtime.h>
#include <hip/hip_bf16.h>

// ---------------------------------------------------------------------------
// Tiny spiking autoencoder scan, persistent-kernel WMMA implementation (v3).
// T=512, B=64, D=1024, H=2048, L=256.  beta=0.9, threshold=1.0.
// v3: WGP-scope prefetch; lat/rec phases use block-level split-K (two waves
//     per 16x16 tile, LDS combine) to halve the dependent WMMA chain depth
//     and double WGP coverage on the long-K phases.
// ---------------------------------------------------------------------------

typedef _Float16 half8 __attribute__((ext_vector_type(8)));
typedef _Float16 v16h  __attribute__((ext_vector_type(16)));
typedef float    v8f   __attribute__((ext_vector_type(8)));

#define T_STEPS 512
#define BB      64
#define DD      1024
#define HH      2048
#define LL      256

// ---- WMMA fragment loaders (contiguous-K order, same for A and B, 16-bit, wave32) -----------
// A (16x32, row-major source, row = lane&15):
//   lanes 0-15 : elems 0..15 = K 0..15
//   lanes 16-31: elems 0..15 = K 16..31
__device__ __forceinline__ v16h loadA_f16(const _Float16* p) {
  half8 lo = *(const half8*)(p);
  half8 hi = *(const half8*)(p + 8);
  return __builtin_shufflevector(lo, hi, 0,1,2,3,4,5,6,7,8,9,10,11,12,13,14,15);
}
// B (32x16, stored N-major so column is contiguous in K; col = lane&15):
//   lanes 0-15 : K 0..15 ; lanes 16-31 : K 16..31
__device__ __forceinline__ v16h loadB_f16(const _Float16* p) {
  half8 lo = *(const half8*)(p);
  half8 hi = *(const half8*)(p + 8);
  return __builtin_shufflevector(lo, hi, 0,1,2,3,4,5,6,7,8,9,10,11,12,13,14,15);
}

// ---- grid-wide barrier (device-scope atomics, all blocks resident) --------
__device__ __forceinline__ void gridBarrier(int* cnt, int* gen, int nwg) {
  __syncthreads();
  if (threadIdx.x == 0) {
    int g = __hip_atomic_load(gen, __ATOMIC_ACQUIRE, __HIP_MEMORY_SCOPE_AGENT);
    int prev = __hip_atomic_fetch_add(cnt, 1, __ATOMIC_ACQ_REL, __HIP_MEMORY_SCOPE_AGENT);
    if (prev == nwg - 1) {
      __hip_atomic_store(cnt, 0, __ATOMIC_RELAXED, __HIP_MEMORY_SCOPE_AGENT);
      __hip_atomic_fetch_add(gen, 1, __ATOMIC_RELEASE, __HIP_MEMORY_SCOPE_AGENT);
    } else {
      while (__hip_atomic_load(gen, __ATOMIC_ACQUIRE, __HIP_MEMORY_SCOPE_AGENT) == g) {
        __builtin_amdgcn_s_sleep(8);
      }
    }
  }
  __syncthreads();
}

__device__ __forceinline__ float waveSum(float v) {
#pragma unroll
  for (int off = 16; off > 0; off >>= 1) v += __shfl_down(v, off, 32);
  return v;
}

// ---- dual-chain WMMA K-loop (register split-K inside one wave) ------------
template <int K>
__device__ __forceinline__ v8f gemmTileF16(const _Float16* ap, const _Float16* bp) {
  v8f c0 = {0.f, 0.f, 0.f, 0.f, 0.f, 0.f, 0.f, 0.f};
  v8f c1 = {0.f, 0.f, 0.f, 0.f, 0.f, 0.f, 0.f, 0.f};
#pragma unroll 2
  for (int k = 0; k < K; k += 64) {
    __builtin_prefetch(bp + k + 1024, 0, 3);   // near-scope prefetch
    v16h a0 = loadA_f16(ap + k);
    v16h b0 = loadB_f16(bp + k);
    v16h a1 = loadA_f16(ap + k + 32);
    v16h b1 = loadB_f16(bp + k + 32);
    c0 = __builtin_amdgcn_wmma_f32_16x16x32_f16(false, a0, false, b0,
                                                (short)0, c0, false, false);
    c1 = __builtin_amdgcn_wmma_f32_16x16x32_f16(false, a1, false, b1,
                                                (short)0, c1, false, false);
  }
  return c0 + c1;
}

// ---- LIF epilogue for one 16x16 tile --------------------------------------
// C/D layout: vgpr r, lanes 0-15 -> M=m0+r, lanes 16-31 -> M=m0+8+r; N=n0+(lane&15)
template <int N, bool WRITE_Z>
__device__ __forceinline__ void lifEpilogue(v8f c, int m0, int n0, int l15,
                                            int hiHalf, const float* bias,
                                            float* V, _Float16* sOut,
                                            float* zOut, float* srAcc,
                                            int lane) {
  const int n = n0 + l15;
  const int mb = m0 + hiHalf * 8;
  const float bn = bias[n];
  float ssum = 0.f;
#pragma unroll
  for (int r = 0; r < 8; ++r) {
    const size_t idx = (size_t)(mb + r) * N + n;
    float v = 0.9f * V[idx] + (c[r] + bn);    // leaky integrate
    const float s = (v > 1.0f) ? 1.0f : 0.0f; // fire (u = v - threshold > 0)
    v -= s;                                   // reset by subtraction
    V[idx] = v;
    sOut[idx] = (_Float16)s;
    if constexpr (WRITE_Z) zOut[idx] = s;
    ssum += s;
  }
  ssum = waveSum(ssum);
  if (lane == 0) atomicAdd(srAcc, ssum);
}

// ---- flat mapping: one tile per wave (enc, dec phases) --------------------
template <int K, int N, bool WRITE_Z>
__device__ __forceinline__ void lifGemm(const _Float16* Ain, const _Float16* Bt,
                                        const float* bias, float* V,
                                        _Float16* sOut, float* zOut,
                                        float* srAcc, int wave, int nWaves,
                                        int lane) {
  constexpr int tilesN = N / 16;
  constexpr int nt = (BB / 16) * tilesN;
  const int l15 = lane & 15;
  const int hiHalf = (lane >> 4) & 1;
  for (int tile = wave; tile < nt; tile += nWaves) {
    const int tm = tile / tilesN;
    const int tn = tile - tm * tilesN;
    const int m0 = tm * 16, n0 = tn * 16;
    const _Float16* ap = Ain + (size_t)(m0 + l15) * K + hiHalf * 16;
    const _Float16* bp = Bt + (size_t)(n0 + l15) * K + hiHalf * 16;
    v8f c = gemmTileF16<K>(ap, bp);
    lifEpilogue<N, WRITE_Z>(c, m0, n0, l15, hiHalf, bias, V, sOut, zOut,
                            srAcc, lane);
  }
}

// ---- block split-K mapping: one tile per block, 2 waves split K -----------
template <int K, int N, bool WRITE_Z>
__device__ __forceinline__ void lifGemmSplit(const _Float16* Ain,
                                             const _Float16* Bt,
                                             const float* bias, float* V,
                                             _Float16* sOut, float* zOut,
                                             float* srAcc, int block,
                                             int nBlocks, int wib, int lane,
                                             float* ldsC) {
  constexpr int tilesN = N / 16;
  constexpr int nt = (BB / 16) * tilesN;
  constexpr int Kh = K / 2;
  const int l15 = lane & 15;
  const int hiHalf = (lane >> 4) & 1;
  for (int tile = block; tile < nt; tile += nBlocks) {
    const int tm = tile / tilesN;
    const int tn = tile - tm * tilesN;
    const int m0 = tm * 16, n0 = tn * 16;
    const _Float16* ap = Ain + (size_t)(m0 + l15) * K + (size_t)wib * Kh + hiHalf * 16;
    const _Float16* bp = Bt + (size_t)(n0 + l15) * K + (size_t)wib * Kh + hiHalf * 16;
    v8f c = gemmTileF16<Kh>(ap, bp);
    if (wib == 1) {
#pragma unroll
      for (int r = 0; r < 8; ++r) ldsC[r * 32 + lane] = c[r];
    }
    __syncthreads();
    if (wib == 0) {
#pragma unroll
      for (int r = 0; r < 8; ++r) c[r] += ldsC[r * 32 + lane];
      lifEpilogue<N, WRITE_Z>(c, m0, n0, l15, hiHalf, bias, V, sOut, zOut,
                              srAcc, lane);
    }
    __syncthreads();
  }
}

// ---- recon phase, block split-K (one tile per block) ----------------------
template <int K, int N>
__device__ __forceinline__ void reconGemmSplit(const _Float16* Ain,
                                               const _Float16* Bt,
                                               const float* bias,
                                               float* reconSeqT,
                                               float* reconAcc,
                                               const float* xT, float* errAcc,
                                               int block, int nBlocks, int wib,
                                               int lane, float* ldsC) {
  constexpr int tilesN = N / 16;
  constexpr int nt = (BB / 16) * tilesN;
  constexpr int Kh = K / 2;
  const int l15 = lane & 15;
  const int hiHalf = (lane >> 4) & 1;
  for (int tile = block; tile < nt; tile += nBlocks) {
    const int tm = tile / tilesN;
    const int tn = tile - tm * tilesN;
    const int m0 = tm * 16, n0 = tn * 16;
    const _Float16* ap = Ain + (size_t)(m0 + l15) * K + (size_t)wib * Kh + hiHalf * 16;
    const _Float16* bp = Bt + (size_t)(n0 + l15) * K + (size_t)wib * Kh + hiHalf * 16;
    v8f c = gemmTileF16<Kh>(ap, bp);
    if (wib == 1) {
#pragma unroll
      for (int r = 0; r < 8; ++r) ldsC[r * 32 + lane] = c[r];
    }
    __syncthreads();
    if (wib == 0) {
      const int n = n0 + l15;
      const int mb = m0 + hiHalf * 8;
      const float bn = bias[n];
      float esum = 0.f;
#pragma unroll
      for (int r = 0; r < 8; ++r) {
        const size_t idx = (size_t)(mb + r) * N + n;
        const float y = c[r] + ldsC[r * 32 + lane] + bn;
        reconSeqT[idx] = y;
        reconAcc[idx] += y;
        esum += fabsf(y - xT[idx]);
      }
      esum = waveSum(esum);
      if (lane == 0) atomicAdd(errAcc, esum);
    }
    __syncthreads();
  }
}

// ---------------------------------------------------------------------------
struct SnnParams {
  const float* x;                                   // [T,B,D] f32
  _Float16* x16;                                    // [B,D] current step, f16
  const float *bEnc, *bLat, *bDec, *bRec;
  const _Float16 *WtE, *WtL, *WtD, *WtR;            // [N,K] f16
  _Float16 *sEnc, *z16, *sDec;                      // activation buffers
  float *Ve, *Vl, *Vd, *reconAcc;                   // LIF state + recon sum
  float *recon, *reconSeq, *latentSeq, *spikeRate, *reconErr;  // outputs
  int *cnt, *gen;                                   // barrier
  float *srAcc, *errAcc;                            // scalar accumulators
};

__global__ __launch_bounds__(64) void snn_persistent(SnnParams p) {
  __shared__ float ldsC[8 * 32];
  const int lane = threadIdx.x & 31;
  const int wib = (int)(threadIdx.x >> 5);          // wave in block (0/1)
  const int wave = (int)((blockIdx.x * blockDim.x + threadIdx.x) >> 5);
  const int nWaves = (int)((gridDim.x * blockDim.x) >> 5);
  const int block = (int)blockIdx.x;
  const int nBlocks = (int)gridDim.x;
  const size_t gtid = (size_t)blockIdx.x * blockDim.x + threadIdx.x;
  const size_t gsz = (size_t)gridDim.x * blockDim.x;

  for (int t = 0; t < T_STEPS; ++t) {
    // encoder: x_t @ W_enc -> s_enc  (512 tiles: one per wave, chain 32)
    lifGemm<DD, HH, false>(p.x16, p.WtE, p.bEnc, p.Ve, p.sEnc, nullptr,
                           p.srAcc + 0, wave, nWaves, lane);
    gridBarrier(p.cnt, p.gen, nBlocks);
    // latent: s_enc @ W_lat -> z_t  (64 tiles: one per block, K split, chain 32)
    lifGemmSplit<HH, LL, true>(p.sEnc, p.WtL, p.bLat, p.Vl, p.z16,
                               p.latentSeq + (size_t)t * BB * LL, p.srAcc + 1,
                               block, nBlocks, wib, lane, ldsC);
    gridBarrier(p.cnt, p.gen, nBlocks);
    // decoder: z_t @ W_dec -> s_dec  (512 tiles: one per wave, chain 8)
    lifGemm<LL, HH, false>(p.z16, p.WtD, p.bDec, p.Vd, p.sDec, nullptr,
                           p.srAcc + 2, wave, nWaves, lane);
    gridBarrier(p.cnt, p.gen, nBlocks);
    // readout: s_dec @ W_rec  (256 tiles: one per block, K split, chain 32)
    reconGemmSplit<HH, DD>(p.sDec, p.WtR, p.bRec,
                           p.reconSeq + (size_t)t * BB * DD, p.reconAcc,
                           p.x + (size_t)t * BB * DD, p.errAcc,
                           block, nBlocks, wib, lane, ldsC);
    // overlap: f32->f16 convert of x_{t+1} for next step's encoder
    if (t + 1 < T_STEPS) {
      const float* xn = p.x + (size_t)(t + 1) * BB * DD;
      for (size_t i = gtid; i < (size_t)BB * DD; i += gsz)
        p.x16[i] = (_Float16)xn[i];
    }
    gridBarrier(p.cnt, p.gen, nBlocks);
  }

  // finalize means
  for (size_t i = gtid; i < (size_t)BB * DD; i += gsz)
    p.recon[i] = p.reconAcc[i] * (1.0f / (float)T_STEPS);
  if (gtid == 0) {
    p.spikeRate[0] = p.srAcc[0] / ((float)T_STEPS * BB * HH);
    p.spikeRate[1] = p.srAcc[1] / ((float)T_STEPS * BB * LL);
    p.spikeRate[2] = p.srAcc[2] / ((float)T_STEPS * BB * HH);
    p.reconErr[0]  = p.errAcc[0] / ((float)T_STEPS * BB * DD);
  }
}

// ---- init: transpose+convert weights to f16 [N,K], x_0 -> f16, zero state -
struct InitParams {
  const float *x, *We, *Wl, *Wd, *Wr;
  _Float16 *x16, *WtE, *WtL, *WtD, *WtR;
  float* zeroBase;      // Ve|Vl|Vd|reconAcc contiguous
  unsigned zeroCount;
  unsigned* hdr;        // cnt, gen, srAcc[3], errAcc
};

__global__ __launch_bounds__(256) void snn_init(InitParams p) {
  const size_t g = (size_t)blockIdx.x * blockDim.x + threadIdx.x;
  const size_t s = (size_t)gridDim.x * blockDim.x;
  if (g < 8) p.hdr[g] = 0u;
  for (size_t i = g; i < (size_t)BB * DD; i += s)
    p.x16[i] = (_Float16)p.x[i];
  // W_enc [D,H] -> WtE [H,D]
  for (size_t i = g; i < (size_t)HH * DD; i += s)
    p.WtE[i] = (_Float16)p.We[(i & (DD - 1)) * HH + (i >> 10)];
  // W_lat [H,L] -> WtL [L,H]
  for (size_t i = g; i < (size_t)LL * HH; i += s)
    p.WtL[i] = (_Float16)p.Wl[(i & (HH - 1)) * LL + (i >> 11)];
  // W_dec [L,H] -> WtD [H,L]
  for (size_t i = g; i < (size_t)HH * LL; i += s)
    p.WtD[i] = (_Float16)p.Wd[(i & (LL - 1)) * HH + (i >> 8)];
  // W_rec [H,D] -> WtR [D,H]
  for (size_t i = g; i < (size_t)DD * HH; i += s)
    p.WtR[i] = (_Float16)p.Wr[(i & (HH - 1)) * DD + (i >> 11)];
  for (size_t i = g; i < p.zeroCount; i += s) p.zeroBase[i] = 0.0f;
}

// ---------------------------------------------------------------------------
extern "C" void kernel_launch(void* const* d_in, const int* in_sizes, int n_in,
                              void* d_out, int out_size, void* d_ws,
                              size_t ws_size, hipStream_t stream) {
  (void)in_sizes; (void)n_in; (void)out_size; (void)ws_size;
  char* ws = (char*)d_ws;
  size_t o = 0;
  auto take = [&](size_t bytes) {
    char* r = ws + o;
    o = (o + bytes + 255) & ~(size_t)255;
    return r;
  };
  unsigned* hdr = (unsigned*)take(256);            // cnt, gen, sr[3], err
  // contiguous zero region: Ve, Vl, Vd, reconAcc
  float* Ve       = (float*)take((size_t)BB * HH * 4);
  float* Vl       = (float*)(ws + o); o += (size_t)BB * LL * 4;
  float* Vd       = (float*)(ws + o); o += (size_t)BB * HH * 4;
  float* reconAcc = (float*)(ws + o); o += (size_t)BB * DD * 4;
  o = (o + 255) & ~(size_t)255;
  const unsigned zeroCount = BB * HH + BB * LL + BB * HH + BB * DD;
  _Float16* WtE  = (_Float16*)take((size_t)HH * DD * 2);
  _Float16* WtL  = (_Float16*)take((size_t)LL * HH * 2);
  _Float16* WtD  = (_Float16*)take((size_t)HH * LL * 2);
  _Float16* WtR  = (_Float16*)take((size_t)DD * HH * 2);
  _Float16* sEnc = (_Float16*)take((size_t)BB * HH * 2);
  _Float16* z16  = (_Float16*)take((size_t)BB * LL * 2);
  _Float16* sDec = (_Float16*)take((size_t)BB * HH * 2);
  _Float16* x16  = (_Float16*)take((size_t)BB * DD * 2);

  float* out = (float*)d_out;
  float* recon     = out;
  float* reconSeq  = out + (size_t)BB * DD;
  float* latentSeq = reconSeq + (size_t)T_STEPS * BB * DD;
  float* spikeRate = latentSeq + (size_t)T_STEPS * BB * LL;
  float* reconErr  = spikeRate + 3;

  InitParams ip;
  ip.x  = (const float*)d_in[0];
  ip.We = (const float*)d_in[1]; ip.Wl = (const float*)d_in[3];
  ip.Wd = (const float*)d_in[5]; ip.Wr = (const float*)d_in[7];
  ip.x16 = x16; ip.WtE = WtE; ip.WtL = WtL; ip.WtD = WtD; ip.WtR = WtR;
  ip.zeroBase = Ve; ip.zeroCount = zeroCount; ip.hdr = hdr;
  snn_init<<<2048, 256, 0, stream>>>(ip);

  SnnParams p;
  p.x    = (const float*)d_in[0];
  p.x16  = x16;
  p.bEnc = (const float*)d_in[2]; p.bLat = (const float*)d_in[4];
  p.bDec = (const float*)d_in[6]; p.bRec = (const float*)d_in[8];
  p.WtE = WtE; p.WtL = WtL; p.WtD = WtD; p.WtR = WtR;
  p.sEnc = sEnc; p.z16 = z16; p.sDec = sDec;
  p.Ve = Ve; p.Vl = Vl; p.Vd = Vd; p.reconAcc = reconAcc;
  p.recon = recon; p.reconSeq = reconSeq; p.latentSeq = latentSeq;
  p.spikeRate = spikeRate; p.reconErr = reconErr;
  p.cnt = (int*)hdr; p.gen = (int*)hdr + 1;
  p.srAcc = (float*)hdr + 2; p.errAcc = (float*)hdr + 5;
  snn_persistent<<<256, 64, 0, stream>>>(p);
}
